// MultiHeadLinear_28200755265768
// MI455X (gfx1250) — 8-bit matrix operand order repaired and checked by reading, compiled, not run on hardware
//
#include <hip/hip_runtime.h>
#include <stdint.h>

#define B_N   4096
#define E_N   8
#define IN_N  256
#define OUT_N 256
#define NPAT  256

typedef __attribute__((ext_vector_type(8))) int v8i;

__global__ __launch_bounds__(256) void k_gate_bin(
    const float* __restrict__ x,
    const float* __restrict__ proj_w,
    const float* __restrict__ proj_b,
    int8_t*      __restrict__ xb,
    int*         __restrict__ pat)
{
    const int tid  = threadIdx.x;
    const int lane = tid & 31;
    const int wave = tid >> 5;
    const int b    = blockIdx.x * 8 + wave;

    float acc[E_N];
#pragma unroll
    for (int e = 0; e < E_N; ++e) acc[e] = 0.f;

#pragma unroll
    for (int j = 0; j < IN_N / 32; ++j) {
        const int i = lane + j * 32;
        const float v = x[b * IN_N + i];
        xb[b * IN_N + i] = (int8_t)((v > 0.f) ? 1 : ((v < 0.f) ? -1 : 0));
#pragma unroll
        for (int e = 0; e < E_N; ++e)
            acc[e] += v * proj_w[e * IN_N + i];
    }

#pragma unroll
    for (int e = 0; e < E_N; ++e) {
        float a = acc[e];
#pragma unroll
        for (int m = 16; m > 0; m >>= 1)
            a += __shfl_xor(a, m, 32);
        acc[e] = a;
    }

    if (lane == 0) {
        int g = 0;
#pragma unroll
        for (int e = 0; e < E_N; ++e)
            if (acc[e] + proj_b[e] > 0.f) g |= (1 << e);
        pat[b] = g;
    }
}

#define WB_STRIDE  256
#define AT_STRIDE  264
#define SMEM_BYTES 65536
#define LIST_OFF   0
#define ATILE_OFF  16384
#define CNT_OFF    (ATILE_OFF + 16 * AT_STRIDE)

__global__ __launch_bounds__(256) void k_moe_bin_gemm(
    const float*  __restrict__ weight,
    const float*  __restrict__ scale,
    const int8_t* __restrict__ xb,
    const int*    __restrict__ pat,
    float*        __restrict__ y)
{
    __shared__ __align__(16) char smem[SMEM_BYTES];
    const int g    = blockIdx.x;
    const int tid  = threadIdx.x;
    const int lane = tid & 31;
    const int wave = tid >> 5;

    {
        const int col = tid;
        char* dst = smem + col * WB_STRIDE;
        for (int i = 0; i < IN_N; i += 4) {
            uint32_t packed = 0;
#pragma unroll
            for (int q = 0; q < 4; ++q) {
                float s = 0.f;
#pragma unroll
                for (int e = 0; e < E_N; ++e)
                    if ((g >> e) & 1)
                        s += weight[(e * IN_N + (i + q)) * OUT_N + col];
                const int8_t sv = (s > 0.f) ? 1 : ((s < 0.f) ? -1 : 0);
                packed |= ((uint32_t)(uint8_t)sv) << (8 * q);
            }
            *(uint32_t*)(dst + i) = packed;
        }
    }
    __syncthreads();

    v8i bfrag[2][4];
#pragma unroll
    for (int t = 0; t < 2; ++t) {
        const int tileN = wave * 2 + t;
        const int col   = tileN * 16 + (lane & 15);
        const char* src = smem + col * WB_STRIDE + ((lane >> 4) << 4);
#pragma unroll
        for (int ks = 0; ks < 4; ++ks) {
#pragma unroll
            for (int j = 0; j < 8; ++j) {
                const int koff = ks * 64 + ((j & 3) << 2) + ((j >> 2) << 5);
                bfrag[t][ks][j] = *(const int*)(src + koff);
            }
        }
    }
    __syncthreads();

    const int   o0  = (wave * 2 + 0) * 16 + (lane & 15);
    const int   o1  = (wave * 2 + 1) * 16 + (lane & 15);
    const float sc0 = scale[o0];
    const float sc1 = scale[o1];

    int*  list  = (int*)(smem + LIST_OFF);
    int*  cnt   = (int*)(smem + CNT_OFF);
    char* atile = smem + ATILE_OFF;

    if (tid == 0) *cnt = 0;
    __syncthreads();
    for (int j = tid; j < B_N; j += 256)
        if (pat[j] == g)
            list[atomicAdd(cnt, 1)] = j;
    __syncthreads();
    const int count = *cnt;
    const int nch   = (count + 15) >> 4;

    for (int c = 0; c < nch; ++c) {
        {
            const int row = tid >> 4;
            const int seg = tid & 15;
            const int idx = c * 16 + row;
            uint32_t v0 = 0, v1 = 0, v2 = 0, v3 = 0;
            if (idx < count) {
                const uint32_t* src =
                    (const uint32_t*)(xb + (size_t)list[idx] * IN_N + seg * 16);
                v0 = src[0]; v1 = src[1]; v2 = src[2]; v3 = src[3];
            }
            uint32_t* dst = (uint32_t*)(atile + row * AT_STRIDE + seg * 16);
            dst[0] = v0; dst[1] = v1; dst[2] = v2; dst[3] = v3;
        }
        __syncthreads();

        int sidx[8];
        {
            const int* lb = list + c * 16 + ((lane >> 4) << 3);
            const int4 la = *(const int4*)(lb);
            const int4 lc = *(const int4*)(lb + 4);
            sidx[0] = la.x; sidx[1] = la.y; sidx[2] = la.z; sidx[3] = la.w;
            sidx[4] = lc.x; sidx[5] = lc.y; sidx[6] = lc.z; sidx[7] = lc.w;
        }

        v8i acc0 = {};
        v8i acc1 = {};
#pragma unroll
        for (int ks = 0; ks < 4; ++ks) {
            v8i a;
            const char* abase =
                atile + (lane & 15) * AT_STRIDE + ks * 64 + ((lane >> 4) << 4);
#pragma unroll
            for (int j = 0; j < 8; ++j) {
                const int off = ((j & 3) << 2) + ((j >> 2) << 5);
                a[j] = *(const int*)(abase + off);
            }
            acc0 = __builtin_amdgcn_wmma_i32_16x16x64_iu8(
                       true, a, true, bfrag[0][ks], acc0, false, false);
            acc1 = __builtin_amdgcn_wmma_i32_16x16x64_iu8(
                       true, a, true, bfrag[1][ks], acc1, false, false);
        }

#pragma unroll
        for (int d = 0; d < 8; ++d) {
            const int row = d + ((lane >> 4) << 3);
            const int idx = c * 16 + row;
            if (idx < count) {
                const size_t s = (size_t)sidx[d];
                y[s * OUT_N + o0] = (float)acc0[d] * sc0;
                y[s * OUT_N + o1] = (float)acc1[d] * sc1;
            }
        }
        __syncthreads();
    }
}

extern "C" void kernel_launch(void* const* d_in, const int* in_sizes, int n_in,
                              void* d_out, int out_size, void* d_ws, size_t ws_size,
                              hipStream_t stream) {
    (void)in_sizes; (void)n_in; (void)out_size; (void)ws_size;
    const float* x      = (const float*)d_in[0];
    const float* weight = (const float*)d_in[1];
    const float* proj_w = (const float*)d_in[2];
    const float* proj_b = (const float*)d_in[3];
    const float* scale  = (const float*)d_in[4];

    int8_t* xb  = (int8_t*)d_ws;
    int*    pat = (int*)((char*)d_ws + (size_t)B_N * IN_N);
    float*  y   = (float*)d_out;

    k_gate_bin<<<B_N / 8, 256, 0, stream>>>(x, proj_w, proj_b, xb, pat);
    k_moe_bin_gemm<<<NPAT, 256, 0, stream>>>(weight, scale, xb, pat, y);
}
